// CodeTransformer_39341900431787
// MI455X (gfx1250) — matrix operand order repaired and checked by reading, not compiled, not run on hardware
//
#include <hip/hip_runtime.h>

// ---------------------------------------------------------------------------
// CDNA5 (gfx1250) bf16 WMMA transformer forward.
// Residual stream fp32; all GEMMs bf16 x bf16 -> f32 via v_wmma_f32_16x16x32_bf16.
// ---------------------------------------------------------------------------

typedef __attribute__((ext_vector_type(16))) __bf16 v16bf;
typedef __attribute__((ext_vector_type(8)))  __bf16 v8bf;
typedef __attribute__((ext_vector_type(4)))  __bf16 v4bf;
typedef __attribute__((ext_vector_type(8)))  float  v8f;

static __device__ __forceinline__ __bf16 f2bf(float f) {
  unsigned u = __float_as_uint(f);
  unsigned short s = (unsigned short)((u + 0x7FFFu + ((u >> 16) & 1u)) >> 16);
  return __builtin_bit_cast(__bf16, s);
}
static __device__ __forceinline__ float bf2f(__bf16 h) {
  unsigned short s = __builtin_bit_cast(unsigned short, h);
  return __uint_as_float(((unsigned)s) << 16);
}

static __device__ __forceinline__ v8f wmma_bf16(v16bf a, v16bf b, v8f c) {
  return __builtin_amdgcn_wmma_f32_16x16x32_bf16(
      /*neg_a=*/false, a, /*neg_b=*/false, b,
      /*c_mod=*/(short)0, c, /*reuse_a=*/false, /*reuse_b=*/false);
}

// A fragment (16x32 bf16, MxK): lane<16 holds row=lane, K chunks {0..7, 16..23};
// lane>=16 holds row=lane-16, K chunks {8..15, 24..31}. Two 16B LDS loads.
static __device__ __forceinline__ v16bf frag_a(const __bf16* base, int stride,
                                               int lane, int koff) {
  int row = lane & 15;
  int c0  = ((lane >> 4) & 1) * 8;
  const __bf16* p = base + row * stride + koff;
  v8bf lo = *(const v8bf*)(p + c0);
  v8bf hi = *(const v8bf*)(p + 16 + c0);
  return __builtin_shufflevector(lo, hi, 0, 1, 2, 3, 4, 5, 6, 7,
                                 8, 9, 10, 11, 12, 13, 14, 15);
}

// B fragment (32x16 bf16, KxN), sourced from an LDS tile stored N-major [N][K]:
// lane<16: col N=lane, K chunks {0..7, 16..23}; lane>=16: col N=lane-16, K chunks {8..15, 24..31}.
static __device__ __forceinline__ v16bf frag_b(const __bf16* base, int stride,
                                               int lane, int koff) {
  int col = lane & 15;
  int kg  = ((lane >> 4) & 1) * 8;
  const __bf16* p = base + col * stride + koff + kg;
  v8bf lo = *(const v8bf*)(p);
  v8bf hi = *(const v8bf*)(p + 16);
  return __builtin_shufflevector(lo, hi, 0, 1, 2, 3, 4, 5, 6, 7,
                                 8, 9, 10, 11, 12, 13, 14, 15);
}

// ---------------------------------------------------------------------------
// Weight prep: fp32 [K][N] -> bf16 [N][K] (transposed), via LDS 32x32 tiles.
// ---------------------------------------------------------------------------
__global__ __launch_bounds__(256) void transpose_cvt_kernel(
    const float* __restrict__ in, __bf16* __restrict__ out, int K, int N) {
  __shared__ float tile[32][33];
  size_t off = (size_t)blockIdx.z * (size_t)K * (size_t)N;
  int k0 = blockIdx.y * 32, n0 = blockIdx.x * 32;
  int tx = threadIdx.x, ty = threadIdx.y;  // blockDim = (32, 8)
  #pragma unroll
  for (int i = 0; i < 32; i += 8)
    tile[ty + i][tx] = in[off + (size_t)(k0 + ty + i) * N + (n0 + tx)];
  __syncthreads();
  #pragma unroll
  for (int i = 0; i < 32; i += 8)
    out[off + (size_t)(n0 + ty + i) * K + (k0 + tx)] = f2bf(tile[tx][ty + i]);
}

// fp32 -> bf16 elementwise (for Whead, already [N][K] for x @ Whead^T).
__global__ __launch_bounds__(256) void cvt_bf16_kernel(
    const float* __restrict__ in, __bf16* __restrict__ out, int n4) {
  int i = blockIdx.x * 256 + threadIdx.x;
  if (i < n4) {
    float4 v = ((const float4*)in)[i];
    v4bf o;
    o[0] = f2bf(v.x); o[1] = f2bf(v.y); o[2] = f2bf(v.z); o[3] = f2bf(v.w);
    ((v4bf*)out)[i] = o;
  }
}

// ---------------------------------------------------------------------------
// Embedding: x[b,s,:] = tok_emb[id] * sqrt(d) + pe[s,:]
// ---------------------------------------------------------------------------
__global__ __launch_bounds__(256) void embed_kernel(
    const int* __restrict__ ids, const float* __restrict__ emb,
    const float* __restrict__ pe, float* __restrict__ x) {
  const int D = 1024, S = 1024;
  int rs = blockIdx.x;          // b*S + s
  int s  = rs & (S - 1);
  int c  = threadIdx.x * 4;
  int id = ids[rs];
  float4 e = *(const float4*)&emb[(size_t)id * D + c];
  float4 p = *(const float4*)&pe[(size_t)s * D + c];
  float4 o = {e.x * 32.0f + p.x, e.y * 32.0f + p.y,
              e.z * 32.0f + p.z, e.w * 32.0f + p.w};
  *(float4*)&x[(size_t)rs * D + c] = o;
}

// ---------------------------------------------------------------------------
// LayerNorm fp32 -> bf16, one 1024-wide row per 256-thread block.
// ---------------------------------------------------------------------------
__global__ __launch_bounds__(256) void layernorm_kernel(
    const float* __restrict__ x, const float* __restrict__ g,
    const float* __restrict__ bta, __bf16* __restrict__ out) {
  const int D = 1024;
  int row = blockIdx.x;
  int tid = threadIdx.x;
  int c = tid * 4;
  float4 v = *(const float4*)&x[(size_t)row * D + c];
  float s  = v.x + v.y + v.z + v.w;
  float sq = v.x * v.x + v.y * v.y + v.z * v.z + v.w * v.w;
  #pragma unroll
  for (int off = 1; off < 32; off <<= 1) {
    s  += __shfl_xor(s, off, 32);
    sq += __shfl_xor(sq, off, 32);
  }
  __shared__ float ss[8], ssq[8];
  int wave = tid >> 5, lane = tid & 31;
  if (lane == 0) { ss[wave] = s; ssq[wave] = sq; }
  __syncthreads();
  float ts = 0.f, tsq = 0.f;
  #pragma unroll
  for (int i = 0; i < 8; ++i) { ts += ss[i]; tsq += ssq[i]; }
  float mu   = ts * (1.0f / 1024.0f);
  float var  = tsq * (1.0f / 1024.0f) - mu * mu;
  float rstd = rsqrtf(var + 1e-5f);
  float4 gg = *(const float4*)&g[c];
  float4 bb = *(const float4*)&bta[c];
  v4bf o;
  o[0] = f2bf((v.x - mu) * rstd * gg.x + bb.x);
  o[1] = f2bf((v.y - mu) * rstd * gg.y + bb.y);
  o[2] = f2bf((v.z - mu) * rstd * gg.z + bb.z);
  o[3] = f2bf((v.w - mu) * rstd * gg.w + bb.w);
  *(v4bf*)&out[(size_t)row * D + c] = o;
}

// ---------------------------------------------------------------------------
// GEMM: C[M,N] = A[M,K](bf16) x Bt[N,K](bf16)  (+bias, +residual, ReLU, out f32/bf16)
// 256 threads = 8 wave32; 128x128x32 tile; 8 WMMA per wave per K-step.
// ---------------------------------------------------------------------------
#define BM 128
#define BN 128
#define BKT 32
#define LDT 40  // padded LDS stride (20-bank skew, conflict-free 16B frags)

template <bool BIAS, bool RES, bool RELU, bool OUTBF>
__global__ __launch_bounds__(256) void gemm_bf16_kernel(
    const __bf16* __restrict__ A, const __bf16* __restrict__ Bt,
    const float* __restrict__ bias, const float* __restrict__ res,
    float* __restrict__ outF, __bf16* __restrict__ outB,
    int M, int N, int K) {
  __shared__ __bf16 As[BM * LDT];
  __shared__ __bf16 Bs[BN * LDT];
  const int tid  = threadIdx.x;
  const int lane = tid & 31;
  const int wave = tid >> 5;
  const int wm = wave & 3;    // 4 waves over M (32 rows each)
  const int wn = wave >> 2;   // 2 waves over N (64 cols each)
  const int bm = blockIdx.y * BM;
  const int bn = blockIdx.x * BN;

  v8f acc[2][4];
  #pragma unroll
  for (int mi = 0; mi < 2; ++mi)
    #pragma unroll
    for (int nj = 0; nj < 4; ++nj)
      #pragma unroll
      for (int e = 0; e < 8; ++e) acc[mi][nj][e] = 0.0f;

  for (int k0 = 0; k0 < K; k0 += BKT) {
    // Cooperative tile load: 512 16B chunks per matrix, 2 per thread.
    #pragma unroll
    for (int c = 0; c < 2; ++c) {
      int chunk = tid + c * 256;
      int row = chunk >> 2;
      int cc  = (chunk & 3) * 8;
      *(v8bf*)&As[row * LDT + cc] = *(const v8bf*)&A[(size_t)(bm + row) * K + k0 + cc];
      *(v8bf*)&Bs[row * LDT + cc] = *(const v8bf*)&Bt[(size_t)(bn + row) * K + k0 + cc];
      if (k0 + BKT < K) {
        __builtin_prefetch(&A[(size_t)(bm + row) * K + k0 + BKT + cc], 0, 1);
        __builtin_prefetch(&Bt[(size_t)(bn + row) * K + k0 + BKT + cc], 0, 1);
      }
    }
    __syncthreads();
    v16bf af[2], bfr[4];
    #pragma unroll
    for (int mi = 0; mi < 2; ++mi)
      af[mi] = frag_a(&As[(wm * 32 + mi * 16) * LDT], LDT, lane, 0);
    #pragma unroll
    for (int nj = 0; nj < 4; ++nj)
      bfr[nj] = frag_b(&Bs[(wn * 64 + nj * 16) * LDT], LDT, lane, 0);
    #pragma unroll
    for (int mi = 0; mi < 2; ++mi)
      #pragma unroll
      for (int nj = 0; nj < 4; ++nj)
        acc[mi][nj] = wmma_bf16(af[mi], bfr[nj], acc[mi][nj]);
    __syncthreads();
  }

  // Epilogue. C layout: vgpr i -> row i (lanes 0-15) / row i+8 (lanes 16-31).
  const int hs = (lane >> 4) & 1;
  const int cl = lane & 15;
  #pragma unroll
  for (int mi = 0; mi < 2; ++mi)
    #pragma unroll
    for (int nj = 0; nj < 4; ++nj)
      #pragma unroll
      for (int i = 0; i < 8; ++i) {
        int row = bm + wm * 32 + mi * 16 + hs * 8 + i;
        int col = bn + wn * 64 + nj * 16 + cl;
        float v = acc[mi][nj][i];
        if (BIAS) v += bias[col];
        if (RES)  v += res[(size_t)row * N + col];
        if (RELU) v = fmaxf(v, 0.0f);
        if (OUTBF) outB[(size_t)row * N + col] = f2bf(v);
        else       outF[(size_t)row * N + col] = v;
      }
}

// ---------------------------------------------------------------------------
// Flash attention: block = (qtile 64 rows) x (b,h); 128 threads = 4 wave32,
// each wave owns a 16-row query stripe. Online softmax, causal, dk = 64.
// ---------------------------------------------------------------------------
__global__ __launch_bounds__(128) void attn_kernel(
    const __bf16* __restrict__ Q, const __bf16* __restrict__ Kb,
    const __bf16* __restrict__ Vb, __bf16* __restrict__ O) {
  const int S = 1024, D = 1024, DK = 64;
  __shared__ __bf16 Qs[64 * 72];
  __shared__ __bf16 Ks[64 * 72];   // [key][dk]   -> B operand for Q.K^T
  __shared__ __bf16 Vt[64 * 72];   // [dk][key]   -> B operand for P.V
  __shared__ __bf16 Ps[64 * 72];   // per-wave 16-row P stripes

  const int tid  = threadIdx.x;
  const int lane = tid & 31;
  const int wave = tid >> 5;
  const int qt = blockIdx.x;              // 16 query tiles
  const int bh = blockIdx.y;              // B*H = 32
  const int b = bh >> 4, h = bh & 15;
  const size_t base = (size_t)b * S * D + (size_t)h * DK;
  const int qbase = qt * 64;

  // Stage the 64x64 Q tile.
  #pragma unroll
  for (int c = 0; c < 4; ++c) {
    int chunk = tid + c * 128;
    int r  = chunk >> 3;
    int cc = (chunk & 7) * 8;
    *(v8bf*)&Qs[r * 72 + cc] = *(const v8bf*)&Q[base + (size_t)(qbase + r) * D + cc];
  }
  __syncthreads();

  v16bf aq0 = frag_a(&Qs[(wave * 16) * 72], 72, lane, 0);
  v16bf aq1 = frag_a(&Qs[(wave * 16) * 72], 72, lane, 32);

  float mrow[8], lrow[8];
  v8f ctxa[4];
  #pragma unroll
  for (int i = 0; i < 8; ++i) { mrow[i] = -1e30f; lrow[i] = 0.0f; }
  #pragma unroll
  for (int j = 0; j < 4; ++j)
    #pragma unroll
    for (int e = 0; e < 8; ++e) ctxa[j][e] = 0.0f;

  const int rl = ((lane >> 4) & 1) * 8;
  const int cl = lane & 15;

  for (int kt = 0; kt <= qt; ++kt) {
    const int kbase = kt * 64;
    // Stage K (row-major) and V (transposed) tiles.
    #pragma unroll
    for (int c = 0; c < 4; ++c) {
      int chunk = tid + c * 128;
      int r  = chunk >> 3;
      int cc = (chunk & 7) * 8;
      *(v8bf*)&Ks[r * 72 + cc] = *(const v8bf*)&Kb[base + (size_t)(kbase + r) * D + cc];
      v8bf vv = *(const v8bf*)&Vb[base + (size_t)(kbase + r) * D + cc];
      #pragma unroll
      for (int j = 0; j < 8; ++j) Vt[(cc + j) * 72 + r] = vv[j];
    }
    __syncthreads();

    // scores = Q . K^T (16q x 64k per wave), K=64 -> 2 chained WMMAs per tile.
    v8f sc[4];
    #pragma unroll
    for (int j = 0; j < 4; ++j) {
      v16bf bk0 = frag_b(&Ks[(j * 16) * 72], 72, lane, 0);
      v16bf bk1 = frag_b(&Ks[(j * 16) * 72], 72, lane, 32);
      v8f z;
      #pragma unroll
      for (int e = 0; e < 8; ++e) z[e] = 0.0f;
      z = wmma_bf16(aq0, bk0, z);
      sc[j] = wmma_bf16(aq1, bk1, z);
    }

    // Scale + causal mask (diagonal tile only).
    const bool diag = (kt == qt);
    const int rb = qbase + wave * 16 + rl;
    #pragma unroll
    for (int j = 0; j < 4; ++j) {
      int col = kbase + j * 16 + cl;
      #pragma unroll
      for (int i = 0; i < 8; ++i) {
        float v = sc[j][i] * 0.125f;  // 1/sqrt(64)
        if (diag && col > rb + i) v = -1e30f;
        sc[j][i] = v;
      }
    }

    // Online softmax (row reductions across 16-lane halves).
    #pragma unroll
    for (int i = 0; i < 8; ++i) {
      float mloc = fmaxf(fmaxf(sc[0][i], sc[1][i]), fmaxf(sc[2][i], sc[3][i]));
      #pragma unroll
      for (int off = 1; off < 16; off <<= 1)
        mloc = fmaxf(mloc, __shfl_xor(mloc, off, 32));
      float mnew = fmaxf(mrow[i], mloc);
      float corr = __expf(mrow[i] - mnew);
      float rs = 0.0f;
      #pragma unroll
      for (int j = 0; j < 4; ++j) {
        float e = __expf(sc[j][i] - mnew);
        sc[j][i] = e;
        rs += e;
      }
      #pragma unroll
      for (int off = 1; off < 16; off <<= 1) rs += __shfl_xor(rs, off, 32);
      lrow[i] = lrow[i] * corr + rs;
      mrow[i] = mnew;
      #pragma unroll
      for (int j = 0; j < 4; ++j) ctxa[j][i] *= corr;
    }

    // P -> LDS (wave-private stripe), then ctx += P.V
    __bf16* pw = &Ps[(wave * 16) * 72];
    #pragma unroll
    for (int j = 0; j < 4; ++j)
      #pragma unroll
      for (int i = 0; i < 8; ++i)
        pw[(rl + i) * 72 + j * 16 + cl] = f2bf(sc[j][i]);
    asm volatile("" ::: "memory");  // same-wave LDS is in-order; block reordering

    v16bf ap0 = frag_a(pw, 72, lane, 0);
    v16bf ap1 = frag_a(pw, 72, lane, 32);
    #pragma unroll
    for (int j = 0; j < 4; ++j) {
      v16bf bv0 = frag_b(&Vt[(j * 16) * 72], 72, lane, 0);
      v16bf bv1 = frag_b(&Vt[(j * 16) * 72], 72, lane, 32);
      ctxa[j] = wmma_bf16(ap0, bv0, ctxa[j]);
      ctxa[j] = wmma_bf16(ap1, bv1, ctxa[j]);
    }
    __syncthreads();  // protect Ks/Vt before next tile's overwrite
  }

  // Normalize and write ctx (bf16, [B,S,d] with head offset).
  #pragma unroll
  for (int j = 0; j < 4; ++j)
    #pragma unroll
    for (int i = 0; i < 8; ++i) {
      int r = qbase + wave * 16 + rl + i;
      int c = j * 16 + cl;
      O[base + (size_t)r * D + c] = f2bf(ctxa[j][i] / lrow[i]);
    }
}

// ---------------------------------------------------------------------------
// Host orchestration
// ---------------------------------------------------------------------------
extern "C" void kernel_launch(void* const* d_in, const int* in_sizes, int n_in,
                              void* d_out, int out_size, void* d_ws, size_t ws_size,
                              hipStream_t stream) {
  const int B = 2, S = 1024, D = 1024, FF = 4096, V = 32000, L = 4;
  const int M = B * S;

  const int*   ids    = (const int*)d_in[0];
  const float* temb   = (const float*)d_in[1];
  const float* pe     = (const float*)d_in[2];
  const float* Wq     = (const float*)d_in[3];
  const float* Wk     = (const float*)d_in[4];
  const float* Wv     = (const float*)d_in[5];
  const float* Wo     = (const float*)d_in[6];
  const float* bo     = (const float*)d_in[7];
  const float* W1     = (const float*)d_in[8];
  const float* b1     = (const float*)d_in[9];
  const float* W2     = (const float*)d_in[10];
  const float* b2     = (const float*)d_in[11];
  const float* ln1g   = (const float*)d_in[12];
  const float* ln1b   = (const float*)d_in[13];
  const float* ln2g   = (const float*)d_in[14];
  const float* ln2b   = (const float*)d_in[15];
  const float* lnfg   = (const float*)d_in[16];
  const float* lnfb   = (const float*)d_in[17];
  const float* Whead  = (const float*)d_in[18];
  float* logits = (float*)d_out;

  // Workspace layout
  char* ws = (char*)d_ws;
  size_t off = 0;
  auto alloc = [&](size_t bytes) { size_t r = off; off += (bytes + 255) & ~(size_t)255; return r; };
  float*  xres  = (float*) (ws + alloc((size_t)M * D * 4));
  __bf16* hbuf  = (__bf16*)(ws + alloc((size_t)M * D * 2));
  __bf16* qb    = (__bf16*)(ws + alloc((size_t)M * D * 2));
  __bf16* kb    = (__bf16*)(ws + alloc((size_t)M * D * 2));
  __bf16* vb    = (__bf16*)(ws + alloc((size_t)M * D * 2));
  __bf16* ctxb  = (__bf16*)(ws + alloc((size_t)M * D * 2));
  __bf16* ffh   = (__bf16*)(ws + alloc((size_t)M * FF * 2));
  __bf16* WqT   = (__bf16*)(ws + alloc((size_t)L * D * D * 2));
  __bf16* WkT   = (__bf16*)(ws + alloc((size_t)L * D * D * 2));
  __bf16* WvT   = (__bf16*)(ws + alloc((size_t)L * D * D * 2));
  __bf16* WoT   = (__bf16*)(ws + alloc((size_t)L * D * D * 2));
  __bf16* W1T   = (__bf16*)(ws + alloc((size_t)L * D * FF * 2));
  __bf16* W2T   = (__bf16*)(ws + alloc((size_t)L * FF * D * 2));
  __bf16* WhB   = (__bf16*)(ws + alloc((size_t)V * D * 2));
  (void)ws_size; (void)n_in; (void)in_sizes; (void)out_size;

  dim3 tb(32, 8);
  // Weight prep: transpose+convert to [N][K] bf16 (all layers via grid.z).
  transpose_cvt_kernel<<<dim3(D / 32, D / 32, L), tb, 0, stream>>>(Wq, WqT, D, D);
  transpose_cvt_kernel<<<dim3(D / 32, D / 32, L), tb, 0, stream>>>(Wk, WkT, D, D);
  transpose_cvt_kernel<<<dim3(D / 32, D / 32, L), tb, 0, stream>>>(Wv, WvT, D, D);
  transpose_cvt_kernel<<<dim3(D / 32, D / 32, L), tb, 0, stream>>>(Wo, WoT, D, D);
  transpose_cvt_kernel<<<dim3(FF / 32, D / 32, L), tb, 0, stream>>>(W1, W1T, D, FF);
  transpose_cvt_kernel<<<dim3(D / 32, FF / 32, L), tb, 0, stream>>>(W2, W2T, FF, D);
  cvt_bf16_kernel<<<(V * D / 4 + 255) / 256, 256, 0, stream>>>(Whead, WhB, V * D / 4);

  embed_kernel<<<M, 256, 0, stream>>>(ids, temb, pe, xres);

  const dim3 gQ(D / BN, M / BM);    // 8 x 16
  const dim3 gF1(FF / BN, M / BM);  // 32 x 16
  const dim3 gHd(V / BN, M / BM);   // 250 x 16

  for (int l = 0; l < L; ++l) {
    layernorm_kernel<<<M, 256, 0, stream>>>(xres, ln1g + l * D, ln1b + l * D, hbuf);
    gemm_bf16_kernel<false, false, false, true><<<gQ, 256, 0, stream>>>(
        hbuf, WqT + (size_t)l * D * D, nullptr, nullptr, nullptr, qb, M, D, D);
    gemm_bf16_kernel<false, false, false, true><<<gQ, 256, 0, stream>>>(
        hbuf, WkT + (size_t)l * D * D, nullptr, nullptr, nullptr, kb, M, D, D);
    gemm_bf16_kernel<false, false, false, true><<<gQ, 256, 0, stream>>>(
        hbuf, WvT + (size_t)l * D * D, nullptr, nullptr, nullptr, vb, M, D, D);
    attn_kernel<<<dim3(S / 64, B * 16), 128, 0, stream>>>(qb, kb, vb, ctxb);
    gemm_bf16_kernel<true, true, false, false><<<gQ, 256, 0, stream>>>(
        ctxb, WoT + (size_t)l * D * D, bo + l * D, xres, xres, nullptr, M, D, D);
    layernorm_kernel<<<M, 256, 0, stream>>>(xres, ln2g + l * D, ln2b + l * D, hbuf);
    gemm_bf16_kernel<true, false, true, true><<<gF1, 256, 0, stream>>>(
        hbuf, W1T + (size_t)l * D * FF, b1 + l * FF, nullptr, nullptr, ffh, M, FF, D);
    gemm_bf16_kernel<true, true, false, false><<<gQ, 256, 0, stream>>>(
        ffh, W2T + (size_t)l * FF * D, b2 + l * D, xres, xres, nullptr, M, D, FF);
  }

  layernorm_kernel<<<M, 256, 0, stream>>>(xres, lnfg, lnfb, hbuf);
  gemm_bf16_kernel<false, false, false, false><<<gHd, 256, 0, stream>>>(
      hbuf, WhB, nullptr, nullptr, logits, nullptr, M, V, D);
}
